// SwinTransformerBlock3D_82643760709814
// MI455X (gfx1250) — matrix operand order repaired and checked by reading, not compiled, not run on hardware
//
#include <hip/hip_runtime.h>
#include <math.h>

typedef _Float16 v16h __attribute__((ext_vector_type(16)));
typedef _Float16 v8h  __attribute__((ext_vector_type(8)));
typedef float    v8f  __attribute__((ext_vector_type(8)));

#define SCALEF 0.17677669529663687f   // 32^-0.5

// ---------------- async-LDS staging (gfx1250) with safe fallback -----------
#if defined(__has_builtin)
#if __has_builtin(__builtin_amdgcn_global_load_async_to_lds_b128)
#define HAVE_ASYNC_LDS 1
#else
#define HAVE_ASYNC_LDS 0
#endif
#else
#define HAVE_ASYNC_LDS 0
#endif

#if HAVE_ASYNC_LDS
// exact pointee type from the toolchain's builtin signature:
// (int __vector_size__(16) __device__*, int __vector_size__(16) __shared__*, Ii, Ii)
typedef int t4i __attribute__((vector_size(16)));
typedef __attribute__((address_space(1))) t4i* as1_t4i_p;
typedef __attribute__((address_space(3))) t4i* as3_t4i_p;
static __device__ __forceinline__ void async_wait0() {
#if __has_builtin(__builtin_amdgcn_s_wait_asynccnt)
  __builtin_amdgcn_s_wait_asynccnt(0);
#else
  asm volatile("s_wait_asynccnt 0x0" ::: "memory");
#endif
}
#endif

// copy n16 * 16 bytes from global gsrc into LDS sdst, block-cooperative
static __device__ __forceinline__ void stage_copy(_Float16* sdst, const _Float16* gsrc,
                                                  int n16, int tid, int nthr) {
#if HAVE_ASYNC_LDS
  for (int i = tid; i < n16; i += nthr) {
    __builtin_amdgcn_global_load_async_to_lds_b128(
        (as1_t4i_p)((const char*)gsrc + (size_t)i * 16),
        (as3_t4i_p)((char*)sdst + (size_t)i * 16), 0, 0);
  }
  async_wait0();
#else
  const uint4* s = (const uint4*)gsrc;
  uint4* d = (uint4*)sdst;
  for (int i = tid; i < n16; i += nthr) d[i] = s[i];
#endif
}

// ---------------- fragment helpers (CDNA5 WMMA 16x16x32 f16 layouts) -------
// A-matrix 16x32 f16: lane L -> row M=L%16; lanes 0-15 K={0..7,16..23},
// lanes 16-31 K={8..15,24..31}.
static __device__ __forceinline__ v16h frag_a(const _Float16* base, int ld) {
  int lane = (int)(threadIdx.x & 31u);
  const _Float16* p = base + (lane & 15) * ld + ((lane >> 4) << 3);
  v8h lo = *(const v8h*)(p);
  v8h hi = *(const v8h*)(p + 16);
  return __builtin_shufflevector(lo, hi, 0,1,2,3,4,5,6,7,8,9,10,11,12,13,14,15);
}
// B-matrix 32x16 f16 (stored as B^T row-major, N rows x K cols):
// lane L -> col N=L%16; lanes 0-15 K={0..7,16..23}, lanes 16-31 K={8..15,24..31}.
static __device__ __forceinline__ v16h frag_b(const _Float16* base, int ld) {
  int lane = (int)(threadIdx.x & 31u);
  const _Float16* p = base + (lane & 15) * ld + ((lane >> 4) << 3);
  v8h lo = *(const v8h*)(p);
  v8h hi = *(const v8h*)(p + 16);
  return __builtin_shufflevector(lo, hi, 0,1,2,3,4,5,6,7,8,9,10,11,12,13,14,15);
}
static __device__ __forceinline__ v8f wmma_f16(v16h a, v16h b, v8f c) {
  return __builtin_amdgcn_wmma_f32_16x16x32_f16(false, a, false, b, (short)0, c, false, false);
}
// wave index as a provably-uniform SGPR value (scalarizes per-wave control flow)
static __device__ __forceinline__ int wave_id() {
  return __builtin_amdgcn_readfirstlane((int)(threadIdx.x >> 5));
}

// ---------------- workspace layout (bytes) ---------------------------------
static constexpr size_t OFF_HBUF  = 0;           // f16 1024*112*128  (dead after qkv)
static constexpr size_t OFF_OBUF  = 0;           // f16 1024*112*128  (aliases hbuf)
static constexpr size_t OFF_G     = 0;           // f16 100352*512    (aliases everything dead by fc1)
static constexpr size_t OFF_Q     = 29360128;    // f16 4096*112*32 (scaled q)
static constexpr size_t OFF_K     = 58720256;    // f16 4096*112*32
static constexpr size_t OFF_VT    = 88080384;    // f16 4096*32*128 (v transposed, keys padded to 128)
static constexpr size_t OFF_Y1    = 121634816;   // f32 100352*128 (x + attn out)
static constexpr size_t OFF_M     = 173015040;   // f16 100352*128 (LN2 out)
static constexpr size_t OFF_QKVT  = 198705152;   // f16 384*128
static constexpr size_t OFF_PROJT = 198803456;   // f16 128*128
static constexpr size_t OFF_FC1T  = 198836224;   // f16 512*128
static constexpr size_t OFF_FC2T  = 198967296;   // f16 128*512
static constexpr size_t OFF_BIAS  = 199098368;   // f32 4*112*112 (rel-pos bias, padded)
static constexpr size_t OFF_MASKP = 199299072;   // f32 512*112*112 (mask, padded, -1e30 key-pad folded)
// total required: 224989184 bytes (~215 MiB)

// ---------------- weight transpose + cast: dst[n*K+k] = src[k*N+n] ---------
__global__ void k_wcast(const float* __restrict__ src, _Float16* __restrict__ dst,
                        int K, int N) {
  int idx = blockIdx.x * 256 + threadIdx.x;
  if (idx >= K * N) return;
  int k = idx / N, n = idx - k * N;
  dst[n * K + k] = (_Float16)src[idx];
}

// ---------------- relative-position bias, padded to 112x112 ----------------
__global__ void k_bias(const float* __restrict__ table, const int* __restrict__ ridx,
                       float* __restrict__ biasb) {
  int idx = blockIdx.x * 256 + threadIdx.x;
  if (idx >= 4 * 112 * 112) return;
  int h = idx / 12544; int rem = idx - h * 12544;
  int r = rem / 112, c = rem - r * 112;
  float v = 0.f;
  if (r < 98 && c < 98) v = table[ridx[r * 98 + c] * 4 + h];
  biasb[idx] = v;
}

// ---------------- shift-window mask, padded, key-pad -inf folded -----------
__global__ void k_maskpad(const float* __restrict__ amask, float* __restrict__ maskP) {
  int idx = blockIdx.x * 256 + threadIdx.x;       // 512*12544
  if (idx >= 512 * 12544) return;
  int wi = idx / 12544; int rem = idx - wi * 12544;
  int r = rem / 112, c = rem - r * 112;
  float v;
  if (c >= 98)      v = -1e30f;                   // pad keys -> softmax zero
  else if (r >= 98) v = 0.f;                      // pad queries (discarded later)
  else              v = amask[(size_t)wi * 9604 + r * 98 + c];
  maskP[idx] = v;
}

// ---------------- LN1 + cyclic shift + window partition -> f16 -------------
__global__ void __launch_bounds__(256) k_ln1(const float* __restrict__ x,
                                             const float* __restrict__ g,
                                             const float* __restrict__ bb,
                                             _Float16* __restrict__ hbuf) {
  int wave = wave_id(), lane = threadIdx.x & 31;
  int tok = blockIdx.x * 8 + wave;                  // 1024*112 tokens
  int w = tok / 112, t = tok - w * 112;
  _Float16* dst = hbuf + (size_t)tok * 128;
  if (t >= 98) { for (int i = 0; i < 4; ++i) dst[i * 32 + lane] = (_Float16)0.f; return; }
  int b  = w >> 9, rem = w & 511;
  int dw = rem >> 6, hw = (rem >> 3) & 7, ww = rem & 7;
  int td = t / 49, rr = t - td * 49, th = rr / 7, tw = rr - th * 7;
  int sd = (dw * 2 + td + 1) & 15;                  // roll(-1) on D
  int sh = hw * 7 + th + 3; if (sh >= 56) sh -= 56; // roll(-3) on H
  int sw = ww * 7 + tw + 3; if (sw >= 56) sw -= 56; // roll(-3) on W
  const float* row = x + ((((size_t)b * 16 + sd) * 56 + sh) * 56 + sw) * 128;
  float v[4], s = 0.f;
  for (int i = 0; i < 4; ++i) { v[i] = row[i * 32 + lane]; s += v[i]; }
  for (int m = 1; m < 32; m <<= 1) s += __shfl_xor(s, m, 32);
  float mu = s * (1.f / 128.f);
  float vs = 0.f;
  for (int i = 0; i < 4; ++i) { float d = v[i] - mu; vs += d * d; }
  for (int m = 1; m < 32; m <<= 1) vs += __shfl_xor(vs, m, 32);
  float inv = rsqrtf(vs * (1.f / 128.f) + 1e-5f);
  for (int i = 0; i < 4; ++i) {
    int c = i * 32 + lane;
    dst[c] = (_Float16)((v[i] - mu) * inv * g[c] + bb[c]);
  }
}

// ---------------- QKV GEMM: per-window 112x384x128, epilogue scatters ------
__global__ void __launch_bounds__(256) k_qkv(const _Float16* __restrict__ hbuf,
                                             const _Float16* __restrict__ qkvT,
                                             const float* __restrict__ qkv_b,
                                             _Float16* __restrict__ qb,
                                             _Float16* __restrict__ kb,
                                             _Float16* __restrict__ vT) {
  __shared__ __align__(16) _Float16 sA[112 * 128];
  int w = blockIdx.x;
  int tid = threadIdx.x, wave = wave_id(), lane = tid & 31;
  int n = lane & 15, half8 = (lane >> 4) << 3;
  __builtin_prefetch(qkvT, 0, 1);
  stage_copy(sA, hbuf + (size_t)w * 112 * 128, 1792, tid, 256);
  __syncthreads();
  v16h bf[3][4];
  for (int j = 0; j < 3; ++j) {
    int nt = wave * 3 + j;
    for (int kt = 0; kt < 4; ++kt) bf[j][kt] = frag_b(qkvT + nt * 16 * 128 + kt * 32, 128);
  }
  for (int mt = 0; mt < 7; ++mt) {
    v16h af[4];
    for (int kt = 0; kt < 4; ++kt) af[kt] = frag_a(sA + mt * 16 * 128 + kt * 32, 128);
    for (int j = 0; j < 3; ++j) {
      v8f acc = {};
      for (int kt = 0; kt < 4; ++kt) acc = wmma_f16(af[kt], bf[j][kt], acc);
      int nt   = wave * 3 + j;       // SGPR
      int part = nt >> 3;            // SGPR -> scalar branch below
      int head = (nt >> 1) & 3;      // SGPR
      int dd   = ((nt & 1) << 4) + n;
      float bias = qkv_b[nt * 16 + n];
      size_t hb = (size_t)(w * 4 + head);
      int t0 = mt * 16 + half8;
      if (part == 2) {
        // v^T: tokens contiguous per lane -> one b128 store
        v8h pack;
        for (int i = 0; i < 8; ++i) pack[i] = (_Float16)(acc[i] + bias);
        *(v8h*)(vT + (hb * 32 + dd) * 128 + t0) = pack;
      } else {
        float mult = part ? 1.f : SCALEF;
        _Float16* dst = (part ? kb : qb) + (hb * 112 + t0) * 32 + dd;
        for (int i = 0; i < 8; ++i) dst[(size_t)i * 32] = (_Float16)((acc[i] + bias) * mult);
      }
    }
  }
  // zero v^T key-pad region t in [112,128) so P@V K-padding is exact
  for (int z = tid; z < 2048; z += 256) {
    int hd = z >> 4, t = 112 + (z & 15);
    vT[((size_t)w * 128 + hd) * 128 + t] = (_Float16)0.f;
  }
}

// ---------------- attention: S=QK^T (+bias+mask) -> softmax -> O=PV --------
__global__ void __launch_bounds__(128) k_attn(const _Float16* __restrict__ qb,
                                              const _Float16* __restrict__ kbuf,
                                              const _Float16* __restrict__ vT,
                                              const float* __restrict__ biasb,
                                              const float* __restrict__ maskP,
                                              _Float16* __restrict__ obuf) {
  __shared__ __align__(16) _Float16 P[112 * 128];   // probs, keys padded to 128
  int bid = blockIdx.x, w = bid >> 2, head = bid & 3;
  int tid = threadIdx.x, wave = wave_id(), lane = tid & 31;
  int n = lane & 15, half8 = (lane >> 4) << 3;
  const _Float16* qbase = qb   + (size_t)bid * 112 * 32;
  const _Float16* kbase = kbuf + (size_t)bid * 112 * 32;
  const _Float16* vbase = vT   + (size_t)bid * 32 * 128;
  const float* biasH = biasb + head * 12544;
  const float* maskW = maskP + (size_t)(w & 511) * 12544;
  int mt_end = (wave * 2 + 2 < 7) ? wave * 2 + 2 : 7;   // SGPR bounds
  for (int mt = wave * 2; mt < mt_end; ++mt) {
    v16h aq = frag_a(qbase + mt * 16 * 32, 32);
    v8f s[7];
    for (int nt = 0; nt < 7; ++nt) {
      v8f z = {};
      s[nt] = wmma_f16(aq, frag_b(kbase + nt * 16 * 32, 32), z);   // K=32 = one WMMA
    }
    for (int i = 0; i < 8; ++i) {
      int row = mt * 16 + half8 + i;
      const float* br = biasH + row * 112 + n;
      const float* mr = maskW + row * 112 + n;
      float rv[7];
      for (int nt = 0; nt < 7; ++nt) rv[nt] = s[nt][i] + br[nt * 16] + mr[nt * 16];
      float m = rv[0];
      for (int nt = 1; nt < 7; ++nt) m = fmaxf(m, rv[nt]);
      for (int d = 1; d < 16; d <<= 1) m = fmaxf(m, __shfl_xor(m, d, 32));
      float sum = 0.f;
      for (int nt = 0; nt < 7; ++nt) { rv[nt] = __expf(rv[nt] - m); sum += rv[nt]; }
      for (int d = 1; d < 16; d <<= 1) sum += __shfl_xor(sum, d, 32);
      float inv = 1.f / sum;
      for (int nt = 0; nt < 7; ++nt) P[row * 128 + nt * 16 + n] = (_Float16)(rv[nt] * inv);
      P[row * 128 + 112 + n] = (_Float16)0.f;       // zero K-pad columns
    }
  }
  __syncthreads();
  for (int mt = wave * 2; mt < mt_end; ++mt) {
    v16h ap[4];
    for (int kt = 0; kt < 4; ++kt) ap[kt] = frag_a(P + mt * 16 * 128 + kt * 32, 128);
    for (int j = 0; j < 2; ++j) {
      v8f acc = {};
      for (int kt = 0; kt < 4; ++kt)
        acc = wmma_f16(ap[kt], frag_b(vbase + j * 16 * 128 + kt * 32, 128), acc);
      int dd = j * 16 + n;
      for (int i = 0; i < 8; ++i) {
        int row = mt * 16 + half8 + i;
        obuf[((size_t)w * 112 + row) * 128 + head * 32 + dd] = (_Float16)acc[i];
      }
    }
  }
}

// ---------------- proj GEMM + window reverse + unshift + residual ----------
__global__ void __launch_bounds__(256) k_proj(const _Float16* __restrict__ obuf,
                                              const _Float16* __restrict__ projT,
                                              const float* __restrict__ proj_b,
                                              const float* __restrict__ x,
                                              float* __restrict__ y1) {
  __shared__ __align__(16) _Float16 sA[112 * 128];
  int w = blockIdx.x;
  int tid = threadIdx.x, wave = wave_id(), lane = tid & 31;
  int n = lane & 15, half8 = (lane >> 4) << 3;
  __builtin_prefetch(projT, 0, 1);
  stage_copy(sA, obuf + (size_t)w * 112 * 128, 1792, tid, 256);
  __syncthreads();
  int nt = wave;
  v16h bf[4];
  for (int kt = 0; kt < 4; ++kt) bf[kt] = frag_b(projT + nt * 16 * 128 + kt * 32, 128);
  int col = nt * 16 + n;
  float pb = proj_b[col];
  int b = w >> 9, rem = w & 511;
  int dw = rem >> 6, hw = (rem >> 3) & 7, ww = rem & 7;
  for (int mt = 0; mt < 7; ++mt) {
    v16h af[4];
    for (int kt = 0; kt < 4; ++kt) af[kt] = frag_a(sA + mt * 16 * 128 + kt * 32, 128);
    v8f acc = {};
    for (int kt = 0; kt < 4; ++kt) acc = wmma_f16(af[kt], bf[kt], acc);
    for (int i = 0; i < 8; ++i) {
      int t = mt * 16 + half8 + i;
      if (t >= 98) continue;                         // drop pad tokens
      int td = t / 49, rr = t - td * 49, th = rr / 7, tw = rr - th * 7;
      int sd = (dw * 2 + td + 1) & 15;
      int sh = hw * 7 + th + 3; if (sh >= 56) sh -= 56;
      int sw = ww * 7 + tw + 3; if (sw >= 56) sw -= 56;
      size_t idx = ((((size_t)b * 16 + sd) * 56 + sh) * 56 + sw) * 128 + col;
      y1[idx] = x[idx] + acc[i] + pb;                // reverse+unshift+residual
    }
  }
}

// ---------------- LN2 -> f16 -----------------------------------------------
__global__ void __launch_bounds__(256) k_ln2(const float* __restrict__ y1,
                                             const float* __restrict__ g,
                                             const float* __restrict__ bb,
                                             _Float16* __restrict__ mbuf) {
  int wave = wave_id(), lane = threadIdx.x & 31;
  int tok = blockIdx.x * 8 + wave;                   // 100352 tokens
  const float* row = y1 + (size_t)tok * 128;
  float v[4], s = 0.f;
  for (int i = 0; i < 4; ++i) { v[i] = row[i * 32 + lane]; s += v[i]; }
  for (int m = 1; m < 32; m <<= 1) s += __shfl_xor(s, m, 32);
  float mu = s * (1.f / 128.f);
  float vs = 0.f;
  for (int i = 0; i < 4; ++i) { float d = v[i] - mu; vs += d * d; }
  for (int m = 1; m < 32; m <<= 1) vs += __shfl_xor(vs, m, 32);
  float inv = rsqrtf(vs * (1.f / 128.f) + 1e-5f);
  _Float16* dst = mbuf + (size_t)tok * 128;
  for (int i = 0; i < 4; ++i) {
    int c = i * 32 + lane;
    dst[c] = (_Float16)((v[i] - mu) * inv * g[c] + bb[c]);
  }
}

// ---------------- FC1 GEMM + exact gelu ------------------------------------
__global__ void __launch_bounds__(256) k_fc1(const _Float16* __restrict__ mbuf,
                                             const _Float16* __restrict__ fc1T,
                                             const float* __restrict__ fc1_b,
                                             _Float16* __restrict__ gbuf) {
  __shared__ __align__(16) _Float16 sA[128 * 128];
  int r0 = blockIdx.x * 128;
  int tid = threadIdx.x, wave = wave_id(), lane = tid & 31;
  int n = lane & 15, half8 = (lane >> 4) << 3;
  __builtin_prefetch(fc1T, 0, 1);
  stage_copy(sA, mbuf + (size_t)r0 * 128, 2048, tid, 256);
  __syncthreads();
  v16h bf[4][4];
  float bias[4];
  for (int j = 0; j < 4; ++j) {
    int nt = wave * 4 + j;
    for (int kt = 0; kt < 4; ++kt) bf[j][kt] = frag_b(fc1T + nt * 16 * 128 + kt * 32, 128);
    bias[j] = fc1_b[nt * 16 + n];
  }
  for (int mt = 0; mt < 8; ++mt) {
    v16h af[4];
    for (int kt = 0; kt < 4; ++kt) af[kt] = frag_a(sA + mt * 16 * 128 + kt * 32, 128);
    for (int j = 0; j < 4; ++j) {
      v8f acc = {};
      for (int kt = 0; kt < 4; ++kt) acc = wmma_f16(af[kt], bf[j][kt], acc);
      int col = (wave * 4 + j) * 16 + n;
      for (int i = 0; i < 8; ++i) {
        int row = r0 + mt * 16 + half8 + i;
        float v = acc[i] + bias[j];
        v = 0.5f * v * (1.f + erff(v * 0.70710678118654752f));   // exact gelu
        gbuf[(size_t)row * 512 + col] = (_Float16)v;
      }
    }
  }
}

// ---------------- FC2 GEMM + final residual -> d_out -----------------------
__global__ void __launch_bounds__(256) k_fc2(const _Float16* __restrict__ gbuf,
                                             const _Float16* __restrict__ fc2T,
                                             const float* __restrict__ fc2_b,
                                             const float* __restrict__ y1,
                                             float* __restrict__ out) {
  int r0 = blockIdx.x * 128;
  int tid = threadIdx.x, wave = wave_id(), lane = tid & 31;
  int n = lane & 15, half8 = (lane >> 4) << 3;
  int nt = wave, col = nt * 16 + n;
  float bias = fc2_b[col];
  __builtin_prefetch(fc2T, 0, 1);
  v16h bf[16];
  for (int kt = 0; kt < 16; ++kt) bf[kt] = frag_b(fc2T + nt * 16 * 512 + kt * 32, 512);
  for (int mt = 0; mt < 8; ++mt) {
    v8f acc = {};
    for (int kt = 0; kt < 16; ++kt) {
      v16h af = frag_a(gbuf + (size_t)(r0 + mt * 16) * 512 + kt * 32, 512);
      acc = wmma_f16(af, bf[kt], acc);
    }
    for (int i = 0; i < 8; ++i) {
      int row = r0 + mt * 16 + half8 + i;
      size_t idx = (size_t)row * 128 + col;
      out[idx] = y1[idx] + acc[i] + bias;
    }
  }
}

// ---------------------------------------------------------------------------
extern "C" void kernel_launch(void* const* d_in, const int* in_sizes, int n_in,
                              void* d_out, int out_size, void* d_ws, size_t ws_size,
                              hipStream_t stream) {
  const float* x      = (const float*)d_in[0];
  const float* amask  = (const float*)d_in[1];
  const int*   ridx   = (const int*)  d_in[2];
  const float* n1g    = (const float*)d_in[3];
  const float* n1b    = (const float*)d_in[4];
  const float* qkv_w  = (const float*)d_in[5];
  const float* qkv_b  = (const float*)d_in[6];
  const float* rtab   = (const float*)d_in[7];
  const float* proj_w = (const float*)d_in[8];
  const float* proj_b = (const float*)d_in[9];
  const float* n2g    = (const float*)d_in[10];
  const float* n2b    = (const float*)d_in[11];
  const float* fc1_w  = (const float*)d_in[12];
  const float* fc1_b  = (const float*)d_in[13];
  const float* fc2_w  = (const float*)d_in[14];
  const float* fc2_b  = (const float*)d_in[15];
  (void)in_sizes; (void)n_in; (void)out_size; (void)ws_size;

  char* ws = (char*)d_ws;
  _Float16* hbuf  = (_Float16*)(ws + OFF_HBUF);
  _Float16* obuf  = (_Float16*)(ws + OFF_OBUF);
  _Float16* gbuf  = (_Float16*)(ws + OFF_G);
  _Float16* qb    = (_Float16*)(ws + OFF_Q);
  _Float16* kb    = (_Float16*)(ws + OFF_K);
  _Float16* vT    = (_Float16*)(ws + OFF_VT);
  float*    y1    = (float*)   (ws + OFF_Y1);
  _Float16* mbuf  = (_Float16*)(ws + OFF_M);
  _Float16* qkvT  = (_Float16*)(ws + OFF_QKVT);
  _Float16* projT = (_Float16*)(ws + OFF_PROJT);
  _Float16* fc1T  = (_Float16*)(ws + OFF_FC1T);
  _Float16* fc2T  = (_Float16*)(ws + OFF_FC2T);
  float*    biasb = (float*)   (ws + OFF_BIAS);
  float*    maskP = (float*)   (ws + OFF_MASKP);

  // weight / bias / mask prep (tiny, L2-resident afterwards)
  k_wcast  <<<192,   256, 0, stream>>>(qkv_w, qkvT, 128, 384);
  k_wcast  <<< 64,   256, 0, stream>>>(proj_w, projT, 128, 128);
  k_wcast  <<<256,   256, 0, stream>>>(fc1_w,  fc1T, 128, 512);
  k_wcast  <<<256,   256, 0, stream>>>(fc2_w,  fc2T, 512, 128);
  k_bias   <<<196,   256, 0, stream>>>(rtab, ridx, biasb);
  k_maskpad<<<25088, 256, 0, stream>>>(amask, maskP);
  // main pipeline
  k_ln1 <<<14336, 256, 0, stream>>>(x, n1g, n1b, hbuf);
  k_qkv <<< 1024, 256, 0, stream>>>(hbuf, qkvT, qkv_b, qb, kb, vT);
  k_attn<<< 4096, 128, 0, stream>>>(qb, kb, vT, biasb, maskP, obuf);
  k_proj<<< 1024, 256, 0, stream>>>(obuf, projT, proj_b, x, y1);
  k_ln2 <<<12544, 256, 0, stream>>>(y1, n2g, n2b, mbuf);
  k_fc1 <<<  784, 256, 0, stream>>>(mbuf, fc1T, fc1_b, gbuf);
  k_fc2 <<<  784, 256, 0, stream>>>(gbuf, fc2T, fc2_b, y1, (float*)d_out);
}
